// GraphMPNN_block_6889127543480
// MI455X (gfx1250) — matrix operand order repaired and checked by reading, not compiled, not run on hardware
//
#include <hip/hip_runtime.h>

typedef __attribute__((ext_vector_type(16))) _Float16 v16h;
typedef __attribute__((ext_vector_type(8)))  _Float16 v8h;
typedef __attribute__((ext_vector_type(8)))  float    v8f;

#define N_TOK 2048          // T*S = 32*64
#define DD    64
#define PANEL 64            // columns staged per LDS panel (== sensor block)
#define NPAN  (N_TOK / PANEL)
#define LN_DECAY -0.3566749439387324f   // ln(0.7)

union V16 { v16h v; v8h h[2]; };

__device__ __forceinline__ v8f wmma_f16(v16h a, v16h b, v8f c) {
  // D = A(16x32 f16) * B(32x16 f16) + C(16x16 f32)
  return __builtin_amdgcn_wmma_f32_16x16x32_f16(false, a, false, b, (short)0, c,
                                                false, false);
}

// Async global->LDS 16B copy (per lane), tracked with ASYNCcnt.
// Generic shared pointers carry the LDS byte offset in their low 32 bits.
__device__ __forceinline__ void async_b128(const void* g, void* l) {
  unsigned loff = (unsigned)(unsigned long long)l;
  asm volatile("global_load_async_to_lds_b128 %0, %1, off"
               :: "v"(loff), "v"(g) : "memory");
}
__device__ __forceinline__ void wait_async0() {
  asm volatile("s_wait_asynccnt 0x0" ::: "memory");
}
__device__ __forceinline__ void wait_ds0() {
  asm volatile("s_wait_dscnt 0x0" ::: "memory");
}

// ---------------------------------------------------------------------------
// Kernel 0: fp16-convert + transpose weights so B-fragments are contiguous.
// ---------------------------------------------------------------------------
__global__ __launch_bounds__(256)
void k_prepw(const float* __restrict__ Wmap, const float* __restrict__ Wth,
             _Float16* __restrict__ WmapT, _Float16* __restrict__ WthT) {
  for (int i = threadIdx.x; i < 64 * 64; i += 256) {
    int k = i >> 6, n = i & 63;
    WmapT[n * 64 + k] = (_Float16)Wmap[k * 64 + n];
    WthT [n * 64 + k] = (_Float16)Wth [k * 64 + n];
  }
}

// ---------------------------------------------------------------------------
// Kernel 1: nf = x @ W_map + b_map (fp16, row-major); also xT fp16.
// ---------------------------------------------------------------------------
__global__ __launch_bounds__(256)
void k_map(const float* __restrict__ x, const float* __restrict__ bmap,
           const _Float16* __restrict__ WmapT,
           _Float16* __restrict__ nfh, _Float16* __restrict__ xTh) {
  const int wave = threadIdx.x >> 5;
  const int lane = threadIdx.x & 31;
  const int lo   = lane & 15;
  const int hi   = (lane >> 4) & 1;
  const int gw   = blockIdx.x * 8 + wave;
  const int b    = gw >> 7;                 // 128 waves per batch
  const int row0 = (gw & 127) << 4;

  const float* xb  = x   + (size_t)b * N_TOK * DD;
  _Float16*   nfb  = nfh + (size_t)b * N_TOK * DD;
  _Float16*   xtb  = xTh + (size_t)b * DD * N_TOK;

  const float* xr = xb + (size_t)(row0 + lo) * DD + (hi ? 16 : 0);
  V16 a0, a1;
  #pragma unroll
  for (int i = 0; i < 8; ++i) {
    a0.v[i]     = (_Float16)xr[i];
    a0.v[8 + i] = (_Float16)xr[8 + i];
    a1.v[i]     = (_Float16)xr[32 + i];
    a1.v[8 + i] = (_Float16)xr[40 + i];
  }

  v8f z = {};
  #pragma unroll
  for (int tt = 0; tt < 4; ++tt) {
    const _Float16* wq = WmapT + (size_t)(16 * tt + lo) * 64 + (hi ? 16 : 0);
    v16h bw0 = *(const v16h*)(wq);
    v16h bw1 = *(const v16h*)(wq + 32);
    v8f c = wmma_f16(a0.v, bw0, z);
    c = wmma_f16(a1.v, bw1, c);
    float bb = bmap[16 * tt + lo];
    #pragma unroll
    for (int e = 0; e < 8; ++e) {
      int row = row0 + e + (hi ? 8 : 0);
      nfb[(size_t)row * DD + 16 * tt + lo] = (_Float16)(c[e] + bb);
    }
  }

  const float* xrow = xb + (size_t)(row0 + lo) * DD;
  #pragma unroll
  for (int i = 0; i < 32; ++i) {
    int d = (hi ? 32 : 0) + i;
    xtb[(size_t)d * N_TOK + row0 + lo] = (_Float16)xrow[d];
  }
}

// ---------------------------------------------------------------------------
// Kernel 2: fused scores -> leaky -> softmax -> decay -> A@X -> theta GEMM.
// Column panels (64 cols) are staged into LDS once per workgroup with async
// global->LDS copies (double buffered); all 8 waves consume them.
// ---------------------------------------------------------------------------
__global__ __launch_bounds__(256)
void k_main(const float* __restrict__ x, const float* __restrict__ bth,
            const _Float16* __restrict__ nfh, const _Float16* __restrict__ xTh,
            const _Float16* __restrict__ WthT, float* __restrict__ out) {
  __shared__ __attribute__((aligned(64))) _Float16 s_nf[2][PANEL * DD]; // 2x8KB
  __shared__ __attribute__((aligned(64))) _Float16 s_xt[2][DD * PANEL]; // 2x8KB
  __shared__ __attribute__((aligned(64))) _Float16 s_p [8][1024];       // 16KB

  const int tid  = threadIdx.x;
  const int wave = tid >> 5;
  const int lane = tid & 31;
  const int lo   = lane & 15;
  const int hi   = (lane >> 4) & 1;
  const int gw   = blockIdx.x * 8 + wave;
  const int b    = gw >> 7;
  const int row0 = (gw & 127) << 4;
  const int trow = row0 >> 6;              // time-block index of this row tile
  _Float16* lp = s_p[wave];

  const _Float16* nfb = nfh + (size_t)b * N_TOK * DD;
  const _Float16* xtb = xTh + (size_t)b * DD * N_TOK;
  const float*    xb  = x   + (size_t)b * N_TOK * DD;

  // Loop-invariant A fragments: this tile's nf rows, K=0..63 in two halves.
  const _Float16* ar = nfb + (size_t)(row0 + lo) * DD + (hi ? 16 : 0);
  V16 a0, a1;
  a0.h[0] = *(const v8h*)(ar);
  a0.h[1] = *(const v8h*)(ar + 8);
  a1.h[0] = *(const v8h*)(ar + 32);
  a1.h[1] = *(const v8h*)(ar + 40);

  v8f z = {};

  // ---------------- pass 1: row max + sum exp of leaky(scores) -------------
  float mx[8], sm[8];
  #pragma unroll
  for (int e = 0; e < 8; ++e) { mx[e] = -3.0e38f; sm[e] = 0.0f; }

  // stage panel 0 (nf rows col0..col0+63 are 8KB contiguous)
  {
    const char* g = (const char*)nfb;
    #pragma unroll
    for (int r = 0; r < 2; ++r) {
      int j = tid + 256 * r;
      async_b128(g + j * 16, (char*)s_nf[0] + j * 16);
    }
  }
  wait_async0();
  __syncthreads();

  for (int p = 0; p < NPAN; ++p) {
    const int col0 = p * PANEL;
    if (p + 1 < NPAN) {
      const char* g = (const char*)(nfb + (size_t)(col0 + PANEL) * DD);
      char* l = (char*)s_nf[(p + 1) & 1];
      #pragma unroll
      for (int r = 0; r < 2; ++r) {
        int j = tid + 256 * r;
        async_b128(g + j * 16, l + j * 16);
      }
    }
    const _Float16* nfp = s_nf[p & 1];
    #pragma unroll
    for (int ct = 0; ct < 4; ++ct) {
      const int colb = col0 + 16 * ct;
      const _Float16* bp = nfp + (size_t)(16 * ct + lo) * DD + (hi ? 16 : 0);
      v16h b0 = *(const v16h*)(bp);
      v16h b1 = *(const v16h*)(bp + 32);
      v8f s = wmma_f16(a0.v, b0, z);
      s = wmma_f16(a1.v, b1, s);
      const int col = colb + lo;
      #pragma unroll
      for (int e = 0; e < 8; ++e) {
        int row = row0 + e + (hi ? 8 : 0);
        float v = s[e];
        if (row == col) v -= 1.0e8f;        // -eye*BIG
        v = v > 0.0f ? v : 0.01f * v;       // leaky_relu
        float nm = fmaxf(mx[e], v);
        sm[e] = sm[e] * __expf(mx[e] - nm) + __expf(v - nm);
        mx[e] = nm;
      }
    }
    wait_async0();
    __syncthreads();
  }
  // 16-lane butterfly merge (a row's 16 columns live across a half-wave).
  #pragma unroll
  for (int off = 1; off < 16; off <<= 1) {
    #pragma unroll
    for (int e = 0; e < 8; ++e) {
      float om = __shfl_xor(mx[e], off, 16);
      float os = __shfl_xor(sm[e], off, 16);
      float nm = fmaxf(mx[e], om);
      sm[e] = sm[e] * __expf(mx[e] - nm) + os * __expf(om - nm);
      mx[e] = nm;
    }
  }
  float inv[8];
  #pragma unroll
  for (int e = 0; e < 8; ++e) inv[e] = 1.0f / sm[e];

  // ---------------- pass 2: acc = (softmax * pre) @ X ----------------------
  v8f acc[4];
  #pragma unroll
  for (int tt = 0; tt < 4; ++tt) acc[tt] = z;

  // stage panel 0 of nf and xT
  {
    const char* g = (const char*)nfb;
    #pragma unroll
    for (int r = 0; r < 2; ++r) {
      int j = tid + 256 * r;
      async_b128(g + j * 16, (char*)s_nf[0] + j * 16);
    }
    #pragma unroll
    for (int r = 0; r < 2; ++r) {
      int j = tid + 256 * r;                // j = d*8 + c16
      int d = j >> 3, c = j & 7;
      async_b128((const char*)(xtb + (size_t)d * N_TOK) + c * 16,
                 (char*)s_xt[0] + j * 16);
    }
  }
  wait_async0();
  __syncthreads();

  for (int p = 0; p < NPAN; ++p) {
    const int col0 = p * PANEL;
    if (p + 1 < NPAN) {
      const char* g = (const char*)(nfb + (size_t)(col0 + PANEL) * DD);
      char* l = (char*)s_nf[(p + 1) & 1];
      char* lx = (char*)s_xt[(p + 1) & 1];
      #pragma unroll
      for (int r = 0; r < 2; ++r) {
        int j = tid + 256 * r;
        async_b128(g + j * 16, l + j * 16);
      }
      #pragma unroll
      for (int r = 0; r < 2; ++r) {
        int j = tid + 256 * r;
        int d = j >> 3, c = j & 7;
        async_b128((const char*)(xtb + (size_t)d * N_TOK + col0 + PANEL) + c * 16,
                   lx + j * 16);
      }
    }
    const _Float16* nfp = s_nf[p & 1];
    const _Float16* xtp = s_xt[p & 1];

    // decay is constant per 64-col panel (panel == sensor block)
    int tc = col0 >> 6;
    int ad = trow > tc ? trow - tc : tc - trow;
    float w = __expf((float)ad * LN_DECAY);

    #pragma unroll
    for (int g2 = 0; g2 < 2; ++g2) {       // two 32-col groups per panel
      const int cg = 32 * g2;
      #pragma unroll
      for (int t = 0; t < 2; ++t) {
        const int colb = col0 + cg + 16 * t;
        const _Float16* bp = nfp + (size_t)(cg + 16 * t + lo) * DD + (hi ? 16 : 0);
        v16h b0 = *(const v16h*)(bp);
        v16h b1 = *(const v16h*)(bp + 32);
        v8f s = wmma_f16(a0.v, b0, z);
        s = wmma_f16(a1.v, b1, s);
        const int col = colb + lo;
        #pragma unroll
        for (int e = 0; e < 8; ++e) {
          int row = row0 + e + (hi ? 8 : 0);
          float v = s[e];
          if (row == col) v -= 1.0e8f;
          v = v > 0.0f ? v : 0.01f * v;
          float pv = __expf(v - mx[e]) * inv[e] * w;
          lp[(e + (hi ? 8 : 0)) * 32 + 16 * t + lo] = (_Float16)pv;
        }
      }
      wait_ds0();
      const _Float16* pr = lp + lo * 32 + (hi ? 16 : 0);
      V16 pa;
      pa.h[0] = *(const v8h*)(pr);
      pa.h[1] = *(const v8h*)(pr + 8);
      #pragma unroll
      for (int tt = 0; tt < 4; ++tt) {
        const _Float16* q = xtp + (size_t)(16 * tt + lo) * PANEL + cg + (hi ? 16 : 0);
        v16h bx = *(const v16h*)(q);
        acc[tt] = wmma_f16(pa.v, bx, acc[tt]);
      }
    }
    wait_async0();
    __syncthreads();
  }

  // eye*pre term: pre_ii = 1  ->  acc += X_row (fp32, exact)
  #pragma unroll
  for (int tt = 0; tt < 4; ++tt)
    #pragma unroll
    for (int e = 0; e < 8; ++e) {
      int row = row0 + e + (hi ? 8 : 0);
      acc[tt][e] += xb[(size_t)row * DD + 16 * tt + lo];
    }

  // Repack acc as fp16 A fragments for the theta GEMM.
  #pragma unroll
  for (int tt = 0; tt < 4; ++tt)
    #pragma unroll
    for (int e = 0; e < 8; ++e)
      lp[(e + (hi ? 8 : 0)) * 64 + 16 * tt + lo] = (_Float16)acc[tt][e];
  wait_ds0();
  const _Float16* fr = lp + lo * 64 + (hi ? 16 : 0);
  V16 f0, f1;
  f0.h[0] = *(const v8h*)(fr);
  f0.h[1] = *(const v8h*)(fr + 8);
  f1.h[0] = *(const v8h*)(fr + 32);
  f1.h[1] = *(const v8h*)(fr + 40);

  float* ob = out + (size_t)b * N_TOK * DD;
  #pragma unroll
  for (int tt = 0; tt < 4; ++tt) {
    const _Float16* wq = WthT + (size_t)(16 * tt + lo) * 64 + (hi ? 16 : 0);
    v16h bw0 = *(const v16h*)(wq);
    v16h bw1 = *(const v16h*)(wq + 32);
    v8f o = wmma_f16(f0.v, bw0, z);
    o = wmma_f16(f1.v, bw1, o);
    float bb = bth[16 * tt + lo];
    #pragma unroll
    for (int e = 0; e < 8; ++e) {
      int row = row0 + e + (hi ? 8 : 0);
      float v = o[e] + bb;
      v = v > 0.0f ? v : 0.01f * v;     // final leaky_relu
      ob[(size_t)row * DD + 16 * tt + lo] = v;
    }
  }
}

// ---------------------------------------------------------------------------
extern "C" void kernel_launch(void* const* d_in, const int* in_sizes, int n_in,
                              void* d_out, int out_size, void* d_ws, size_t ws_size,
                              hipStream_t stream) {
  const float* x    = (const float*)d_in[0];
  const float* Wmap = (const float*)d_in[1];
  const float* bmap = (const float*)d_in[2];
  const float* Wth  = (const float*)d_in[3];
  const float* bth  = (const float*)d_in[4];
  float* out = (float*)d_out;

  // Workspace layout (fp16): WmapT | WthT | nfh | xTh  (~4.02 MB total)
  _Float16* WmapT = (_Float16*)d_ws;
  _Float16* WthT  = WmapT + 64 * 64;
  _Float16* nfh   = WthT + 64 * 64;
  _Float16* xTh   = nfh + (size_t)8 * N_TOK * DD;

  k_prepw<<<1, 256, 0, stream>>>(Wmap, Wth, WmapT, WthT);
  // 8 batches * 2048 rows / 16 rows-per-wave / 8 waves-per-block = 128 blocks
  k_map <<<128, 256, 0, stream>>>(x, bmap, WmapT, nfh, xTh);
  k_main<<<128, 256, 0, stream>>>(x, bth, nfh, xTh, WthT, out);
}
